// Self_Space_Region_Attention_12163347382560
// MI455X (gfx1250) — matrix operand order repaired and checked by reading, not compiled, not run on hardware
//
#include <hip/hip_runtime.h>

// ---------------------------------------------------------------------------
// CDNA5 (gfx1250) implementation: all large GEMMs via v_wmma_f32_16x16x32_bf16.
// Layouts chosen so every WMMA fragment is a contiguous 16B global load:
//   A-fragment: row-major [M,K]  (lane m = L&15, two 8-elem chunks at k+hi*8, k+16+hi*8)
//   B-fragment: stored transposed, row-major [N,K] (lane n = L&15, two 8-elem chunks at k+hi*8, k+16+hi*8)
//   C/D tile:   lane col n = L&15, VGPR i -> row m0 + (L>>4)*8 + i
// ---------------------------------------------------------------------------

typedef __bf16 bf16;
typedef __bf16  bf16x8  __attribute__((ext_vector_type(8)));
typedef __bf16  bf16x16 __attribute__((ext_vector_type(16)));
typedef float   f32x8   __attribute__((ext_vector_type(8)));

static __device__ __forceinline__ bf16x16 frag_cat(bf16x8 lo, bf16x8 hi) {
  bf16x16 r;
#pragma unroll
  for (int i = 0; i < 8; ++i) { r[i] = lo[i]; r[i + 8] = hi[i]; }
  return r;
}

// A tile (16x32 bf16) at base pointer "A" (row-major, leading dim lda elems).
static __device__ __forceinline__ bf16x16 load_a_frag(const bf16* __restrict__ A,
                                                      int lda, int lane) {
  int lr = lane & 15, hi = lane >> 4;
  const bf16* p = A + (size_t)lr * lda + hi * 8;
  bf16x8 lo = *(const bf16x8*)(p);
  bf16x8 hh = *(const bf16x8*)(p + 16);
  return frag_cat(lo, hh);
}

// B tile (32x16 bf16) given B-transposed storage "BT" (row-major [16 x 32], ld ldbt).
static __device__ __forceinline__ bf16x16 load_b_frag(const bf16* __restrict__ BT,
                                                      int ldbt, int lane) {
  int lc = lane & 15, hi = lane >> 4;
  const bf16* p = BT + (size_t)lc * ldbt + hi * 8;
  bf16x8 lo = *(const bf16x8*)(p);
  bf16x8 hh = *(const bf16x8*)(p + 16);
  return frag_cat(lo, hh);
}

static __device__ __forceinline__ f32x8 wmma_bf16(bf16x16 a, bf16x16 b, f32x8 c) {
  return __builtin_amdgcn_wmma_f32_16x16x32_bf16(false, a, false, b, (short)0, c,
                                                 false, false);
}

// ---------------------------------------------------------------------------
// Kernel 0: elementwise f32 -> bf16 (weights)
// ---------------------------------------------------------------------------
__global__ void k_f32_to_bf16(const float* __restrict__ in, bf16* __restrict__ out, int n) {
  int i = blockIdx.x * blockDim.x + threadIdx.x;
  if (i < n) out[i] = (bf16)in[i];
}

// ---------------------------------------------------------------------------
// Kernel 1: x [B][256][4096] f32 -> xT [B][4096][256] bf16 (LDS tile transpose)
// ---------------------------------------------------------------------------
__global__ void k_transpose_x(const float* __restrict__ x, bf16* __restrict__ xT) {
  __shared__ float tile[32][33];
  int b = blockIdx.z;
  int n0 = blockIdx.x * 32, c0 = blockIdx.y * 32;
  const float* xb = x + (size_t)b * 256 * 4096;
  bf16* xTb = xT + (size_t)b * 4096 * 256;
  int tx = threadIdx.x, ty = threadIdx.y;  // 32 x 8
#pragma unroll
  for (int j = 0; j < 4; ++j)
    tile[ty + j * 8][tx] = xb[(size_t)(c0 + ty + j * 8) * 4096 + n0 + tx];
  __syncthreads();
#pragma unroll
  for (int j = 0; j < 4; ++j)
    xTb[(size_t)(n0 + ty + j * 8) * 256 + c0 + tx] = (bf16)tile[tx][ty + j * 8];
}

// ---------------------------------------------------------------------------
// Kernel 2: generic WMMA GEMM, bf16 out + bias.
// out[b][m][n] = sum_k A[m][k]*BT[b][n][k] + bias[m]
// block = 256 threads (8 waves, 2x4), tile 64(M) x 128(N), K multiple of 32.
// ---------------------------------------------------------------------------
__global__ void k_gemm_bf16out(const bf16* __restrict__ A, const bf16* __restrict__ BT,
                               const float* __restrict__ bias, bf16* __restrict__ out,
                               int N, int K, size_t btBatchStride, size_t outBatchStride) {
  int b = blockIdx.z;
  const bf16* BTb = BT + (size_t)b * btBatchStride;
  bf16* outb = out + (size_t)b * outBatchStride;
  int lane = threadIdx.x & 31, wave = threadIdx.x >> 5;
  int wm = wave & 1, wn = wave >> 1;
  int m0 = blockIdx.y * 64 + wm * 32;
  int n0 = blockIdx.x * 128 + wn * 32;
  f32x8 acc[2][2] = {};
  for (int k0 = 0; k0 < K; k0 += 32) {
    bf16x16 a0 = load_a_frag(A + (size_t)m0 * K + k0, K, lane);
    bf16x16 a1 = load_a_frag(A + (size_t)(m0 + 16) * K + k0, K, lane);
    bf16x16 b0 = load_b_frag(BTb + (size_t)n0 * K + k0, K, lane);
    bf16x16 b1 = load_b_frag(BTb + (size_t)(n0 + 16) * K + k0, K, lane);
    acc[0][0] = wmma_bf16(a0, b0, acc[0][0]);
    acc[0][1] = wmma_bf16(a0, b1, acc[0][1]);
    acc[1][0] = wmma_bf16(a1, b0, acc[1][0]);
    acc[1][1] = wmma_bf16(a1, b1, acc[1][1]);
  }
  int lc = lane & 15, hi = lane >> 4;
#pragma unroll
  for (int mt = 0; mt < 2; ++mt)
#pragma unroll
    for (int nt = 0; nt < 2; ++nt)
#pragma unroll
      for (int i = 0; i < 8; ++i) {
        int row = m0 + mt * 16 + hi * 8 + i;
        int col = n0 + nt * 16 + lc;
        outb[(size_t)row * N + col] = (bf16)(acc[mt][nt][i] + bias[row]);
      }
}

// ---------------------------------------------------------------------------
// Kernel 3: softmax over the 1024-channel axis, in place (bf16), per (b,n).
// Also emits RqT [b][h][n][64] (transposed Rq) for the expansion GEMM.
// ---------------------------------------------------------------------------
__global__ void k_softmax_regions(bf16* __restrict__ r, bf16* __restrict__ RqT) {
  int b = blockIdx.y;
  int n = blockIdx.x * blockDim.x + threadIdx.x;  // 0..4095
  bf16* rb = r + (size_t)b * 1024 * 4096;
  float m = -3.4e38f, s = 0.f;
  for (int o = 0; o < 1024; ++o) {
    float v = (float)rb[(size_t)o * 4096 + n];
    float nm = fmaxf(m, v);
    s = s * __expf(m - nm) + __expf(v - nm);
    m = nm;
  }
  float inv = 1.f / s;
  for (int o = 0; o < 1024; ++o) {
    size_t idx = (size_t)o * 4096 + n;
    float p = __expf((float)rb[idx] - m) * inv;
    bf16 pb = (bf16)p;
    rb[idx] = pb;
    int h = o >> 7, j = o & 127;  // channel = h*128 + j ; j<64 -> Rq
    if (j < 64)
      RqT[(((size_t)b * 8 + h) * 4096 + n) * 64 + j] = pb;
  }
}

// ---------------------------------------------------------------------------
// Kernel 4: region pooling, K-split for parallelism.
// Per (b,h,p,seg): [32 x 512] x [512 x 64] WMMA partial GEMM.
// p=0: qr = q^T Rq ; p=1: kr = k^T Rk ; p=2: vr = v^T Rk.
// partial out: part [seg][p][bh][32][64] f32.
// block = 128 threads (4 waves), wave w owns r-columns w*16..w*16+15.
// ---------------------------------------------------------------------------
#define POOL_SEGS 8
#define POOL_SEGK (4096 / POOL_SEGS)

__global__ void k_pool(const bf16* __restrict__ qkv, const bf16* __restrict__ rsm,
                       float* __restrict__ part) {
  int bh = blockIdx.x;            // b*8 + h
  int p = blockIdx.y;             // 0=q,1=k,2=v
  int seg = blockIdx.z;           // K segment
  int b = bh >> 3, h = bh & 7;
  int lane = threadIdx.x & 31, wave = threadIdx.x >> 5;
  int sel = p * 32;               // q/k/v row offset inside the 96-row head block
  int roff = (p == 0) ? 0 : 64;   // Rq vs Rk inside the 128-row head block
  const bf16* Abase = qkv + ((size_t)b * 768 + h * 96 + sel) * 4096;     // [32][4096]
  const bf16* BTbase = rsm + ((size_t)b * 1024 + h * 128 + roff) * 4096; // [64][4096]
  int r0 = wave * 16;
  int kbeg = seg * POOL_SEGK, kend = kbeg + POOL_SEGK;
  f32x8 acc[2] = {};
  for (int k0 = kbeg; k0 < kend; k0 += 32) {
    bf16x16 bfr = load_b_frag(BTbase + (size_t)r0 * 4096 + k0, 4096, lane);
    bf16x16 a0 = load_a_frag(Abase + k0, 4096, lane);
    bf16x16 a1 = load_a_frag(Abase + (size_t)16 * 4096 + k0, 4096, lane);
    acc[0] = wmma_bf16(a0, bfr, acc[0]);
    acc[1] = wmma_bf16(a1, bfr, acc[1]);
  }
  float* outp = part + (((size_t)seg * 3 + p) * 128 + bh) * 2048;
  int lc = lane & 15, hi = lane >> 4;
#pragma unroll
  for (int mt = 0; mt < 2; ++mt)
#pragma unroll
    for (int i = 0; i < 8; ++i) {
      int row = mt * 16 + hi * 8 + i;
      outp[row * 64 + r0 + lc] = acc[mt][i];
    }
}

// Deterministic segment reduction: pool[i] = sum_seg part[seg][i]
__global__ void k_pool_reduce(const float* __restrict__ part, float* __restrict__ pool) {
  int i = blockIdx.x * blockDim.x + threadIdx.x;  // 3*128*2048 total
  float s = 0.f;
#pragma unroll
  for (int seg = 0; seg < POOL_SEGS; ++seg)
    s += part[(size_t)seg * 3 * 128 * 2048 + i];
  pool[i] = s;
}

// ---------------------------------------------------------------------------
// Kernel 5: tiny region attention per (b,h): 64x64 logits, softmax, vals[d][q].
// ~33 MFLOP total -> plain VALU + LDS. Output bf16 [bh][32][64] (A of expansion).
// ---------------------------------------------------------------------------
__global__ void k_region_attn(const float* __restrict__ pool, bf16* __restrict__ vals) {
  __shared__ float qr_s[2048], kr_s[2048], vr_s[2048], at_s[4096];
  int bh = blockIdx.x;
  int t = threadIdx.x;  // 64 threads
  const float* qr = pool + (size_t)bh * 2048;
  const float* kr = pool + (size_t)(128 + bh) * 2048;
  const float* vr = pool + (size_t)(256 + bh) * 2048;
  for (int i = t; i < 2048; i += 64) { qr_s[i] = qr[i]; kr_s[i] = kr[i]; vr_s[i] = vr[i]; }
  __syncthreads();
  {
    int q = t;
    float row[64];
    float mx = -3.4e38f;
    for (int kk = 0; kk < 64; ++kk) {
      float s = 0.f;
      for (int dd = 0; dd < 32; ++dd) s += qr_s[dd * 64 + q] * kr_s[dd * 64 + kk];
      s *= 0.17677669529663687f;  // 1/sqrt(head_dim=32)
      row[kk] = s;
      mx = fmaxf(mx, s);
    }
    float sum = 0.f;
    for (int kk = 0; kk < 64; ++kk) { row[kk] = __expf(row[kk] - mx); sum += row[kk]; }
    float inv = 1.f / sum;
    for (int kk = 0; kk < 64; ++kk) at_s[q * 64 + kk] = row[kk] * inv;
  }
  __syncthreads();
  bf16* vout = vals + (size_t)bh * 2048;
  for (int idx = t; idx < 2048; idx += 64) {
    int d = idx >> 6, q = idx & 63;
    float s = 0.f;
    for (int kk = 0; kk < 64; ++kk) s += vr_s[d * 64 + kk] * at_s[q * 64 + kk];
    vout[d * 64 + q] = (bf16)s;
  }
}

// ---------------------------------------------------------------------------
// Kernel 6: expansion. Per (b,h): [32 x 64] x [64 x 4096] WMMA GEMM,
// D stored TRANSPOSED into valsT [b][n][e] bf16 (e = h*32 + d), so each lane's
// 8 accumulator rows become one contiguous 16B bf16x8 store.
// block = 256 (8 waves), wave owns one 16-col n-tile.
// ---------------------------------------------------------------------------
__global__ void k_expand(const bf16* __restrict__ vals, const bf16* __restrict__ RqT,
                         bf16* __restrict__ valsT) {
  int bh = blockIdx.x;
  int b = bh >> 3, h = bh & 7;
  int lane = threadIdx.x & 31, wave = threadIdx.x >> 5;
  int n0 = (blockIdx.y * 8 + wave) * 16;
  const bf16* A = vals + (size_t)bh * 2048;             // [32][64]
  const bf16* BT = RqT + (size_t)bh * 4096 * 64;        // [4096][64]
  f32x8 acc[2] = {};
#pragma unroll
  for (int k0 = 0; k0 < 64; k0 += 32) {
    bf16x16 bfr = load_b_frag(BT + (size_t)n0 * 64 + k0, 64, lane);
    bf16x16 a0 = load_a_frag(A + k0, 64, lane);
    bf16x16 a1 = load_a_frag(A + 16 * 64 + k0, 64, lane);
    acc[0] = wmma_bf16(a0, bfr, acc[0]);
    acc[1] = wmma_bf16(a1, bfr, acc[1]);
  }
  int lc = lane & 15, hi = lane >> 4;
#pragma unroll
  for (int mt = 0; mt < 2; ++mt) {
    bf16x8 v;
#pragma unroll
    for (int i = 0; i < 8; ++i) v[i] = (bf16)acc[mt][i];
    // lane holds rows m = mt*16 + hi*8 + {0..7} (contiguous), column n = n0+lc
    *(bf16x8*)(valsT + (((size_t)b * 4096) + n0 + lc) * 256 + h * 32 + mt * 16 + hi * 8) = v;
  }
}

// ---------------------------------------------------------------------------
// Kernel 7: output GEMM + fused epilogue: out = x + alpha*(Wout @ vals + bout)
// M=256, K=256, N=4096 per batch; B^T = valsT [n][e].
// ---------------------------------------------------------------------------
__global__ void k_final(const bf16* __restrict__ Wo, const bf16* __restrict__ valsT,
                        const float* __restrict__ bout, const float* __restrict__ x,
                        const float* __restrict__ alpha, float* __restrict__ out) {
  int b = blockIdx.z;
  const bf16* BTb = valsT + (size_t)b * 4096 * 256;
  const float* xb = x + (size_t)b * 256 * 4096;
  float* outb = out + (size_t)b * 256 * 4096;
  int lane = threadIdx.x & 31, wave = threadIdx.x >> 5;
  int wm = wave & 1, wn = wave >> 1;
  int m0 = blockIdx.y * 64 + wm * 32;
  int n0 = blockIdx.x * 128 + wn * 32;
  f32x8 acc[2][2] = {};
  for (int k0 = 0; k0 < 256; k0 += 32) {
    bf16x16 a0 = load_a_frag(Wo + (size_t)m0 * 256 + k0, 256, lane);
    bf16x16 a1 = load_a_frag(Wo + (size_t)(m0 + 16) * 256 + k0, 256, lane);
    bf16x16 b0 = load_b_frag(BTb + (size_t)n0 * 256 + k0, 256, lane);
    bf16x16 b1 = load_b_frag(BTb + (size_t)(n0 + 16) * 256 + k0, 256, lane);
    acc[0][0] = wmma_bf16(a0, b0, acc[0][0]);
    acc[0][1] = wmma_bf16(a0, b1, acc[0][1]);
    acc[1][0] = wmma_bf16(a1, b0, acc[1][0]);
    acc[1][1] = wmma_bf16(a1, b1, acc[1][1]);
  }
  float al = alpha[0];
  int lc = lane & 15, hi = lane >> 4;
#pragma unroll
  for (int mt = 0; mt < 2; ++mt)
#pragma unroll
    for (int nt = 0; nt < 2; ++nt)
#pragma unroll
      for (int i = 0; i < 8; ++i) {
        int row = m0 + mt * 16 + hi * 8 + i;
        int col = n0 + nt * 16 + lc;
        size_t idx = (size_t)row * 4096 + col;
        outb[idx] = xb[idx] + al * (acc[mt][nt][i] + bout[row]);
      }
}

// ---------------------------------------------------------------------------
extern "C" void kernel_launch(void* const* d_in, const int* in_sizes, int n_in,
                              void* d_out, int out_size, void* d_ws, size_t ws_size,
                              hipStream_t stream) {
  (void)in_sizes; (void)n_in; (void)out_size; (void)ws_size;
  const float* x     = (const float*)d_in[0];
  const float* Wqkv  = (const float*)d_in[1];
  const float* bqkv  = (const float*)d_in[2];
  const float* Wr    = (const float*)d_in[3];
  const float* br    = (const float*)d_in[4];
  const float* Wout  = (const float*)d_in[5];
  const float* bout  = (const float*)d_in[6];
  const float* alpha = (const float*)d_in[7];
  float* out = (float*)d_out;

  char* ws = (char*)d_ws;
  size_t off = 0;
  auto take = [&](size_t bytes) -> void* {
    void* p = ws + off;
    off = (off + bytes + 255) & ~(size_t)255;
    return p;
  };
  bf16*  xT    = (bf16*)take((size_t)16 * 4096 * 256 * 2);       // 32 MiB
  bf16*  Wq_b  = (bf16*)take((size_t)768 * 256 * 2);
  bf16*  Wr_b  = (bf16*)take((size_t)1024 * 256 * 2);
  bf16*  Wo_b  = (bf16*)take((size_t)256 * 256 * 2);
  bf16*  qkv   = (bf16*)take((size_t)16 * 768 * 4096 * 2);       // 96 MiB
  bf16*  rbuf  = (bf16*)take((size_t)16 * 1024 * 4096 * 2);      // 128 MiB (in-place softmax)
  bf16*  RqT   = (bf16*)take((size_t)16 * 8 * 4096 * 64 * 2);    // 64 MiB
  float* part  = (float*)take((size_t)POOL_SEGS * 3 * 128 * 2048 * 4); // 24 MiB
  float* pool  = (float*)take((size_t)3 * 128 * 2048 * 4);       // 3 MiB
  bf16*  vals  = (bf16*)take((size_t)128 * 2048 * 2);            // 0.5 MiB
  bf16*  valsT = (bf16*)take((size_t)16 * 4096 * 256 * 2);       // 32 MiB

  // weights -> bf16
  k_f32_to_bf16<<<(768 * 256 + 255) / 256, 256, 0, stream>>>(Wqkv, Wq_b, 768 * 256);
  k_f32_to_bf16<<<(1024 * 256 + 255) / 256, 256, 0, stream>>>(Wr, Wr_b, 1024 * 256);
  k_f32_to_bf16<<<(256 * 256 + 255) / 256, 256, 0, stream>>>(Wout, Wo_b, 256 * 256);
  // x -> xT bf16
  k_transpose_x<<<dim3(128, 8, 16), dim3(32, 8), 0, stream>>>(x, xT);
  // qkv = Wqkv @ x  (M=768), r = Wr @ x (M=1024)
  k_gemm_bf16out<<<dim3(32, 12, 16), 256, 0, stream>>>(
      Wq_b, xT, bqkv, qkv, 4096, 256, (size_t)4096 * 256, (size_t)768 * 4096);
  k_gemm_bf16out<<<dim3(32, 16, 16), 256, 0, stream>>>(
      Wr_b, xT, br, rbuf, 4096, 256, (size_t)4096 * 256, (size_t)1024 * 4096);
  // channel softmax (in place) + RqT
  k_softmax_regions<<<dim3(16, 16), 256, 0, stream>>>(rbuf, RqT);
  // region pooling qr/kr/vr: K-split partials + deterministic reduce
  k_pool<<<dim3(128, 3, POOL_SEGS), 128, 0, stream>>>(qkv, rbuf, part);
  k_pool_reduce<<<(3 * 128 * 2048) / 256, 256, 0, stream>>>(part, pool);
  // region attention core
  k_region_attn<<<128, 64, 0, stream>>>(pool, vals);
  // expand back to tokens, transposed store
  k_expand<<<dim3(128, 32), 256, 0, stream>>>(vals, RqT, valsT);
  // out = x + alpha * (Wout @ vals + bout)
  k_final<<<dim3(32, 4, 16), 256, 0, stream>>>(Wo_b, valsT, bout, x, alpha, out);
}
